// Int8LinearWeightOnly_43018392437442
// MI455X (gfx1250) — compile-verified, matrix operand order checked by reading, not run on hardware
//
#include <hip/hip_runtime.h>
#include <cstdint>
#include <cstddef>

// Problem constants (from the reference)
#define K_DIM 4096     // IN_FEATURES
#define O_DIM 11008    // OUT_FEATURES (= 172 blocks of 64)
#define B_DIM 64       // BATCH        (= 4 tiles of 16)

#define NCHUNK (K_DIM / 64)   // 64 K-chunks of 64
#define ROWPAD 72             // LDS row stride (bank-conflict-free, 8B aligned)
#define PLANE  (64 * ROWPAD)  // 4608 B per 64x64 plane
#define BUFSZ  (3 * PLANE)    // Ah | Al | Bw per buffer

typedef int v8i __attribute__((ext_vector_type(8)));
typedef int v2i __attribute__((ext_vector_type(2)));
using lds_i8  = __attribute__((address_space(3))) int8_t;
using lds_v2i = __attribute__((address_space(3))) v2i;

// ---------------------------------------------------------------------------
// Kernel 1: per-row 16-bit fixed-point quantization of x into two int8 planes
//   x[b,i] ~= sb[b] * (256*qhi[b,i] + (uint8)qlo[b,i])
// ---------------------------------------------------------------------------
__global__ void __launch_bounds__(256)
quantize_x_kernel(const float* __restrict__ x,
                  int8_t* __restrict__ qhi,
                  int8_t* __restrict__ qlo,
                  float*  __restrict__ sb) {
    __shared__ float red[256];
    const int row = blockIdx.x;
    const int t   = threadIdx.x;
    const float* xr = x + (size_t)row * K_DIM;

    float vals[K_DIM / 256];
    float amax = 0.0f;
    #pragma unroll
    for (int j = 0; j < K_DIM / 256; ++j) {
        float v = xr[t + j * 256];
        vals[j] = v;
        amax = fmaxf(amax, fabsf(v));
    }
    red[t] = amax;
    __syncthreads();
    for (int s = 128; s > 0; s >>= 1) {
        if (t < s) red[t] = fmaxf(red[t], red[t + s]);
        __syncthreads();
    }
    const float rmax = red[0];
    const float sc   = (rmax > 0.0f) ? (rmax / 32767.0f) : 1.0f;
    const float inv  = 1.0f / sc;
    if (t == 0) sb[row] = sc;

    #pragma unroll
    for (int j = 0; j < K_DIM / 256; ++j) {
        int q = (int)rintf(vals[j] * inv);
        q = q > 32767 ? 32767 : (q < -32767 ? -32767 : q);
        const size_t idx = (size_t)row * K_DIM + t + j * 256;
        qhi[idx] = (int8_t)(q >> 8);      // signed high byte
        qlo[idx] = (int8_t)(q & 0xFF);    // unsigned low byte
    }
}

// IU8 fragment: per-lane K bytes {kc+0..7, +16..23, +32..39, +48..55} of one
// padded LDS row. Lowers to ds_load_2addr_b64 pairs.
static __device__ __forceinline__ v8i lds_frag8(const lds_i8* p) {
    v2i c0 = *(const lds_v2i*)(p);
    v2i c1 = *(const lds_v2i*)(p + 16);
    v2i c2 = *(const lds_v2i*)(p + 32);
    v2i c3 = *(const lds_v2i*)(p + 48);
    v8i v;
    v[0] = c0[0]; v[1] = c0[1];
    v[2] = c1[0]; v[3] = c1[1];
    v[4] = c2[0]; v[5] = c2[1];
    v[6] = c3[0]; v[7] = c3[1];
    return v;
}

// One async 16B global->LDS transfer per lane (LDS offset in VDST VGPR).
static __device__ __forceinline__ void async_copy16(unsigned lds_off, uint64_t gaddr) {
    asm volatile("global_load_async_to_lds_b128 %0, %1, off"
                 :: "v"(lds_off), "v"(gaddr) : "memory");
}

static __device__ __forceinline__ void wait_async_and_barrier() {
    asm volatile("s_wait_asynccnt 0x0" ::: "memory");
    __syncthreads();
}

// ---------------------------------------------------------------------------
// Kernel 2: block computes 64(M) x 64(N); 8 wave32 waves, each owns
// (mtile = w&3) x 2 N-tiles (npair = w>>2). K loop: 64 chunks of 64,
// explicitly unrolled x2 over the LDS double buffer so each buffer's body is
// symmetric and accumulators keep fixed homes.
// ---------------------------------------------------------------------------
__global__ void __launch_bounds__(256)
int8_linear_iu8_wmma_kernel(const int8_t* __restrict__ qhi,   // [64][4096]
                            const int8_t* __restrict__ qlo,   // [64][4096]
                            const float*  __restrict__ sb,    // [64]
                            const int8_t* __restrict__ wq,    // [11008][4096]
                            const float*  __restrict__ scale, // [11008]
                            const float*  __restrict__ bias,  // [11008]
                            float*        __restrict__ out) { // [64][11008]
    __shared__ __align__(16) int8_t smem[2 * BUFSZ];

    // Launder the LDS base so the async-asm "memory" clobbers are seen as
    // potential writers of smem (otherwise the ds_loads fold to undef).
    unsigned sbase = (unsigned)(uintptr_t)(lds_i8*)smem;
    asm volatile("" : "+v"(sbase));
    lds_i8* sp = (lds_i8*)(uintptr_t)sbase;

    const int t     = threadIdx.x;
    const int wave  = t >> 5;
    const int lane  = t & 31;
    const int lm    = lane & 15;
    const int hi    = lane >> 4;
    const int mtile = wave & 3;
    const int npair = wave >> 2;
    const int mbase = mtile << 4;
    const int nbase = blockIdx.x * 64;

    // --- staging geometry: thread t moves 16B per plane per chunk ---
    const int srow = t >> 2;            // 0..63
    const int scol = (t & 3) << 4;      // 0,16,32,48
    const unsigned lds_rc = sbase + (unsigned)(srow * ROWPAD + scol);
    const uint64_t g_off  = (uint64_t)srow * K_DIM + (unsigned)scol;
    const uint64_t ga_h   = (uint64_t)(uintptr_t)qhi + g_off;
    const uint64_t ga_l   = (uint64_t)(uintptr_t)qlo + g_off;
    const uint64_t ga_b   = (uint64_t)(uintptr_t)wq +
                            (uint64_t)(nbase + srow) * K_DIM + (unsigned)scol;

    // --- compute-side LDS fragment offsets (within one buffer) ---
    const int kc = hi << 3;
    const unsigned aoff  = (unsigned)((mbase + lm) * ROWPAD + kc);
    const unsigned b0off = (unsigned)(2 * PLANE + ((npair * 2 + 0) * 16 + lm) * ROWPAD + kc);
    const unsigned b1off = (unsigned)(2 * PLANE + ((npair * 2 + 1) * 16 + lm) * ROWPAD + kc);

    v8i ch0 = {}, cl0 = {}, ch1 = {}, cl1 = {};   // exact int32 accumulators

    // stage chunk at byte offset kb into buffer buf (0/1)
    auto stage = [&](int buf, uint64_t kb) {
        const unsigned base = (unsigned)buf * BUFSZ;
        async_copy16(base + lds_rc,             ga_h + kb);
        async_copy16(base + PLANE + lds_rc,     ga_l + kb);
        async_copy16(base + 2 * PLANE + lds_rc, ga_b + kb);
    };

    auto compute = [&](int buf) {
        lds_i8* b = sp + buf * BUFSZ;
        v8i ah = lds_frag8(b + aoff);
        v8i al = lds_frag8(b + PLANE + aoff);
        v8i b0 = lds_frag8(b + b0off);
        v8i b1 = lds_frag8(b + b1off);
        // (sgn_a, A, sgn_b, B, C, reuse_a, reuse_b)
        ch0 = __builtin_amdgcn_wmma_i32_16x16x64_iu8(true,  ah, true, b0, ch0, false, false);
        cl0 = __builtin_amdgcn_wmma_i32_16x16x64_iu8(false, al, true, b0, cl0, false, false);
        ch1 = __builtin_amdgcn_wmma_i32_16x16x64_iu8(true,  ah, true, b1, ch1, false, false);
        cl1 = __builtin_amdgcn_wmma_i32_16x16x64_iu8(false, al, true, b1, cl1, false, false);
    };

    stage(0, 0);
    #pragma unroll 1
    for (int c = 0; c < NCHUNK; c += 2) {
        // ---- buffer 0 holds chunk c ----
        wait_async_and_barrier();
        stage(1, (uint64_t)(c + 1) << 6);           // prefetch chunk c+1
        compute(0);

        // ---- buffer 1 holds chunk c+1 ----
        wait_async_and_barrier();
        if (c + 2 < NCHUNK) stage(0, (uint64_t)(c + 2) << 6);
        compute(1);
    }

    // Epilogue: D element (M = r + 8*hi, N = lm) per N-tile
    #pragma unroll
    for (int cmb = 0; cmb < 2; ++cmb) {
        const int o  = nbase + (npair * 2 + cmb) * 16 + lm;
        const float so = scale[o];
        const float bo = bias[o];
        const v8i ph = cmb ? ch1 : ch0;
        const v8i pl = cmb ? cl1 : cl0;
        #pragma unroll
        for (int r = 0; r < 8; ++r) {
            const int row = mbase + r + (hi << 3);
            const float comb = fmaf(256.0f, (float)ph[r], (float)pl[r]);
            out[(size_t)row * O_DIM + o] = fmaf(comb, sb[row] * so, bo);
        }
    }
}

// ---------------------------------------------------------------------------
// Launch. d_ws layout: qhi (256 KB) | qlo (256 KB) | sb (256 B)
// ---------------------------------------------------------------------------
extern "C" void kernel_launch(void* const* d_in, const int* in_sizes, int n_in,
                              void* d_out, int out_size, void* d_ws, size_t ws_size,
                              hipStream_t stream) {
    const float*  x     = (const float*) d_in[0];
    const int8_t* wq    = (const int8_t*)d_in[1];
    const float*  scale = (const float*) d_in[2];
    const float*  bias  = (const float*) d_in[3];
    float*        out   = (float*)d_out;

    int8_t* qhi = (int8_t*)d_ws;
    int8_t* qlo = qhi + (size_t)B_DIM * K_DIM;
    float*  sb  = (float*)(qlo + (size_t)B_DIM * K_DIM);

    quantize_x_kernel<<<B_DIM, 256, 0, stream>>>(x, qhi, qlo, sb);

    int8_linear_iu8_wmma_kernel<<<O_DIM / 64, 256, 0, stream>>>(
        qhi, qlo, sb, wq, scale, bias, out);
}
